// DMSD_59725815218870
// MI455X (gfx1250) — matrix operand order repaired and checked by reading, not compiled, not run on hardware
//
#include <hip/hip_runtime.h>
#include <hip/hip_bf16.h>
#include <math.h>

// ---------------------------------------------------------------------------
// BERT-ish cross-attention stack for MI455X (gfx1250), bf16 WMMA everywhere.
//   L=3, B=16, S1=S2=512, D=768, H=12, DH=64, FF=3072
// GEMM: C = A[M,K] * Bt[N,K]^T (both bf16 row-major), fp32 accumulation via
// v_wmma_f32_16x16x32_bf16. Each wave owns a 64x32 tile (4x2 WMMA frags),
// block = 8 waves (4 in M, 2 in N) = 256x64 tile. Fragments are loaded
// directly with b128 global loads off 6 per-lane stream pointers that are
// advanced once per 4 k-steps, so the hot loop is immediate-offset loads +
// WMMAs only (no per-iteration 64-bit VALU address math, no WMMA->VALU WAR
// hazard NOPs). Operand reuse comes from L0/L2 (192 MB L2 holds the whole
// per-layer GEMM working set). Streaming fp32 tensors (scores, probs) use
// non-temporal cache hints so they don't evict it.
// ---------------------------------------------------------------------------

#define L_  3
#define H_  12
#define D_  768
#define DH_ 64
#define FF_ 3072
#define B_  16
#define S1_ 512
#define S2_ 512

typedef __attribute__((ext_vector_type(16))) __bf16 v16bf;
typedef __attribute__((ext_vector_type(8)))  float  v8f;

union FragU { v16bf v; uint4 q[2]; };

__device__ __forceinline__ unsigned short f2bf(float f) {
  unsigned int u = __float_as_uint(f);
  u += 0x7FFFu + ((u >> 16) & 1u);   // round-to-nearest-even
  return (unsigned short)(u >> 16);
}

// Two contiguous 16B loads -> one 16x32 (A) or 32x16 (B) bf16 fragment.
// A (lane m=lane&15, hi=lane>>4): a[e] = A[m][k0+(e&7)+8*hi+16*(e>>3)]
//   -> chunks at +0 and +16 from (row base + k0 + 8*hi).
// B from Bt[N,K] (lane n=lane&15): b[e] = Bt[n][k0+(e&7)+8*hi+16*(e>>3)]
//   -> chunks at +0 and +16 from (col base + k0 + 8*hi).
__device__ __forceinline__ v16bf ldfrag(const unsigned short* __restrict__ p, int o1) {
  FragU u;
  u.q[0] = *(const uint4*)(p);
  u.q[1] = *(const uint4*)(p + o1);
  return u.v;
}

// epi flags:
//   1  = store fp32 to Cf
//   2  = store bf16 to Cb
//   4  = exact-erf gelu before store
//   8  = store bf16 TRANSPOSED: Cb[col*ldcT + row]  (produces V^T directly)
//   16 = fp32 store is non-temporal (streaming tensor, bypass L2 retention)
//   32 = fused block reduction of sum/sumsq into redOut[0..1] (float atomics)
__global__ void __launch_bounds__(256)
wmma_gemm_kernel(const unsigned short* __restrict__ A,
                 const unsigned short* __restrict__ Bt,
                 const float* __restrict__ bias,
                 float* __restrict__ Cf, unsigned short* __restrict__ Cb,
                 int M, int N, int K, int lda, int ldb, int ldc,
                 long aSB, long aSH, long bSB, long bSH, long cSB, long cSH,
                 int nH, float scale, int epi, int ldcT,
                 float* __restrict__ redOut) {
  const int z  = blockIdx.z;
  const int bb = z / nH;
  const int hh = z - bb * nH;
  const unsigned short* Ab = A + (long)bb * aSB + (long)hh * aSH;
  const unsigned short* Bb = Bt + (long)bb * bSB + (long)hh * bSH;
  const long cOff = (long)bb * cSB + (long)hh * cSH;

  const int wave = threadIdx.x >> 5;
  const int lane = threadIdx.x & 31;
  const int ln   = lane & 15;
  const int hi   = lane >> 4;
  const int m0   = blockIdx.y * 256 + (wave >> 1) * 64;   // 4 waves in M
  const int n0   = blockIdx.x * 64  + (wave & 1) * 32;    // 2 waves in N

  v8f c[4][2] = {};
  if (m0 + 64 <= M && n0 + 32 <= N) {       // always true for our shapes
    // 6 per-lane stream pointers; the k-loop only bumps these by constants.
    const unsigned short* pa0 = Ab + (long)(m0 + ln) * lda + 8 * hi;
    const unsigned short* pa1 = pa0 + 16L * lda;
    const unsigned short* pa2 = pa0 + 32L * lda;
    const unsigned short* pa3 = pa0 + 48L * lda;
    const unsigned short* pb0 = Bb + (long)(n0 + ln) * ldb + 8 * hi;
    const unsigned short* pb1 = pb0 + 16L * ldb;

    auto step = [&](int kk) {   // kk is a literal -> immediate load offsets
      v16bf a0 = ldfrag(pa0 + kk, 16);
      v16bf a1 = ldfrag(pa1 + kk, 16);
      v16bf a2 = ldfrag(pa2 + kk, 16);
      v16bf a3 = ldfrag(pa3 + kk, 16);
      v16bf b0 = ldfrag(pb0 + kk, 16);
      v16bf b1 = ldfrag(pb1 + kk, 16);
      c[0][0] = __builtin_amdgcn_wmma_f32_16x16x32_bf16(false, a0, false, b0, (short)0, c[0][0], false, false);
      c[0][1] = __builtin_amdgcn_wmma_f32_16x16x32_bf16(false, a0, false, b1, (short)0, c[0][1], false, false);
      c[1][0] = __builtin_amdgcn_wmma_f32_16x16x32_bf16(false, a1, false, b0, (short)0, c[1][0], false, false);
      c[1][1] = __builtin_amdgcn_wmma_f32_16x16x32_bf16(false, a1, false, b1, (short)0, c[1][1], false, false);
      c[2][0] = __builtin_amdgcn_wmma_f32_16x16x32_bf16(false, a2, false, b0, (short)0, c[2][0], false, false);
      c[2][1] = __builtin_amdgcn_wmma_f32_16x16x32_bf16(false, a2, false, b1, (short)0, c[2][1], false, false);
      c[3][0] = __builtin_amdgcn_wmma_f32_16x16x32_bf16(false, a3, false, b0, (short)0, c[3][0], false, false);
      c[3][1] = __builtin_amdgcn_wmma_f32_16x16x32_bf16(false, a3, false, b1, (short)0, c[3][1], false, false);
    };

    int k0 = 0;
    for (; k0 + 128 <= K; k0 += 128) {       // main: 4 k-steps, 1 ptr bump
      step(0); step(32); step(64); step(96);
      pa0 += 128; pa1 += 128; pa2 += 128; pa3 += 128;
      pb0 += 128; pb1 += 128;
    }
    for (; k0 < K; k0 += 32) {               // remainder (K=64 case)
      step(0);
      pa0 += 32; pa1 += 32; pa2 += 32; pa3 += 32;
      pb0 += 32; pb1 += 32;
    }

    // C/D layout: lane (hi?16:0)+n holds C[v + 8*hi][n] in element v.
    float rs = 0.0f, rss = 0.0f;
    #pragma unroll
    for (int i = 0; i < 4; ++i) {
      #pragma unroll
      for (int j = 0; j < 2; ++j) {
        const int col = n0 + j * 16 + ln;
        const float bv = bias ? bias[col] : 0.0f;
        #pragma unroll
        for (int v = 0; v < 8; ++v) {
          const int row = m0 + i * 16 + hi * 8 + v;
          float val = c[i][j][v] * scale + bv;
          if (epi & 4) val = 0.5f * val * (1.0f + erff(val * 0.70710678118654752f));
          if (epi & 32) { rs += val; rss += val * val; }
          const long idx = cOff + (long)row * ldc + col;
          if (epi & 1) {
            if (epi & 16) __builtin_nontemporal_store(val, Cf + idx);
            else          Cf[idx] = val;
          }
          if (epi & 2) Cb[idx] = f2bf(val);
          if (epi & 8) Cb[(long)col * ldcT + row] = f2bf(val);
        }
      }
    }

    if (epi & 32) {   // fused global-moments reduction (saves a 201MB re-read)
      __shared__ float sh[512];
      sh[threadIdx.x]       = rs;
      sh[256 + threadIdx.x] = rss;
      __syncthreads();
      for (int w = 128; w > 0; w >>= 1) {
        if ((int)threadIdx.x < w) {
          sh[threadIdx.x]       += sh[threadIdx.x + w];
          sh[256 + threadIdx.x] += sh[256 + threadIdx.x + w];
        }
        __syncthreads();
      }
      if (threadIdx.x == 0) {
        atomicAdd(&redOut[0], sh[0]);
        atomicAdd(&redOut[1], sh[256]);
      }
    }
  }
}

// ---------------------------------------------------------------------------

__global__ void cvt_f32_bf16_kernel(const float* __restrict__ in,
                                    unsigned short* __restrict__ out, long n) {
  long i = (long)blockIdx.x * blockDim.x + threadIdx.x;
  if (i < n) out[i] = f2bf(in[i]);
}

// in: [z][K][N] fp32  ->  out: [z][N][K] bf16 (LDS tiled transpose)
__global__ void transpose_cvt_kernel(const float* __restrict__ in,
                                     unsigned short* __restrict__ out,
                                     int K, int N) {
  __shared__ float tile[32][33];
  const long off = (long)blockIdx.z * K * N;
  const int k0 = blockIdx.y * 32, n0 = blockIdx.x * 32;
  const int tx = threadIdx.x, ty = threadIdx.y;     // (32,8)
  #pragma unroll
  for (int r = 0; r < 32; r += 8)
    tile[ty + r][tx] = in[off + (long)(k0 + ty + r) * N + n0 + tx];
  __syncthreads();
  #pragma unroll
  for (int r = 0; r < 32; r += 8)
    out[off + (long)(n0 + ty + r) * K + k0 + tx] = f2bf(tile[tx][ty + r]);
}

__global__ void zero2_kernel(float* p) { if (threadIdx.x < 2) p[threadIdx.x] = 0.0f; }

// one block per score row of 512: z-normalize with GLOBAL mean/std, softmax,
// emit probs fp32 (d_out, NT) + bf16 (workspace, feeds ctx GEMM).
__global__ void __launch_bounds__(256)
softmax_kernel(const float* __restrict__ scores, const float* __restrict__ red,
               float cnt, float* __restrict__ probsF,
               unsigned short* __restrict__ probsB) {
  const float s = red[0], ss = red[1];
  const float mean = s / cnt;
  const float var  = fmaxf((ss - s * s / cnt) / (cnt - 1.0f), 1e-30f); // unbiased
  const float istd = rsqrtf(var);
  const long base = (long)blockIdx.x * S2_;
  const int t = threadIdx.x;
  float v0 = (__builtin_nontemporal_load(scores + base + t)       - mean) * istd;
  float v1 = (__builtin_nontemporal_load(scores + base + t + 256) - mean) * istd;
  __shared__ float sh[256];
  sh[t] = fmaxf(v0, v1); __syncthreads();
  for (int w = 128; w > 0; w >>= 1) { if (t < w) sh[t] = fmaxf(sh[t], sh[t + w]); __syncthreads(); }
  const float rmax = sh[0]; __syncthreads();
  const float e0 = expf(v0 - rmax), e1 = expf(v1 - rmax);
  sh[t] = e0 + e1; __syncthreads();
  for (int w = 128; w > 0; w >>= 1) { if (t < w) sh[t] += sh[t + w]; __syncthreads(); }
  const float inv = 1.0f / sh[0];
  const float p0 = e0 * inv, p1 = e1 * inv;
  __builtin_nontemporal_store(p0, probsF + base + t);
  __builtin_nontemporal_store(p1, probsF + base + t + 256);
  probsB[base + t]       = f2bf(p0);
  probsB[base + t + 256] = f2bf(p1);
}

// out = LN(y + resid) with gamma/beta; one block (256 thr) per row of 768.
__global__ void __launch_bounds__(256)
ln_residual_kernel(const float* __restrict__ y, const float* __restrict__ resid,
                   const float* __restrict__ gamma, const float* __restrict__ beta,
                   float* __restrict__ outF, unsigned short* __restrict__ outB) {
  const long row = blockIdx.x;
  const float* yp = y + row * D_;
  const float* rp = resid + row * D_;
  const int t = threadIdx.x;
  float x0 = yp[t]       + rp[t];
  float x1 = yp[t + 256] + rp[t + 256];
  float x2 = yp[t + 512] + rp[t + 512];
  __shared__ float sh[256];
  sh[t] = x0 + x1 + x2; __syncthreads();
  for (int w = 128; w > 0; w >>= 1) { if (t < w) sh[t] += sh[t + w]; __syncthreads(); }
  const float mean = sh[0] * (1.0f / D_); __syncthreads();
  const float d0 = x0 - mean, d1 = x1 - mean, d2 = x2 - mean;
  sh[t] = d0 * d0 + d1 * d1 + d2 * d2; __syncthreads();
  for (int w = 128; w > 0; w >>= 1) { if (t < w) sh[t] += sh[t + w]; __syncthreads(); }
  const float istd = rsqrtf(sh[0] * (1.0f / D_) + 1e-12f);
  const long base = row * D_;
  const float dv[3] = {d0, d1, d2};
  #pragma unroll
  for (int i = 0; i < 3; ++i) {
    const int col = t + i * 256;
    const float o = gamma[col] * dv[i] * istd + beta[col];
    outF[base + col] = o;
    outB[base + col] = f2bf(o);
  }
}

// ---------------------------------------------------------------------------

extern "C" void kernel_launch(void* const* d_in, const int* in_sizes, int n_in,
                              void* d_out, int out_size, void* d_ws, size_t ws_size,
                              hipStream_t stream) {
  (void)in_sizes; (void)n_in; (void)out_size; (void)ws_size;
  const float* Xin = (const float*)d_in[0];
  const float* Yin = (const float*)d_in[1];
  const float* Wq  = (const float*)d_in[3];  const float* bq = (const float*)d_in[4];
  const float* Wk  = (const float*)d_in[5];  const float* bk = (const float*)d_in[6];
  const float* Wv  = (const float*)d_in[7];  const float* bv = (const float*)d_in[8];
  const float* Wo  = (const float*)d_in[9];  const float* bo = (const float*)d_in[10];
  const float* g1  = (const float*)d_in[11]; const float* be1 = (const float*)d_in[12];
  const float* Wi  = (const float*)d_in[13]; const float* bi = (const float*)d_in[14];
  const float* Wf  = (const float*)d_in[15]; const float* bf = (const float*)d_in[16];
  const float* g2  = (const float*)d_in[17]; const float* be2 = (const float*)d_in[18];

  const long MX = (long)B_ * S1_;                 // 8192 rows
  const long NS = (long)B_ * H_ * S1_ * S2_;      // 50331648 score elements

  // ---- workspace carve-up (bf16 buffers are unsigned short) ----
  char* wsb = (char*)d_ws; size_t cur = 0;
  auto alloc = [&](size_t bytes) -> void* {
    void* p = wsb + cur; cur += (bytes + 255) & ~(size_t)255; return p;
  };
  unsigned short* Xb     = (unsigned short*)alloc(MX * D_ * 2);
  unsigned short* Yb     = (unsigned short*)alloc(MX * D_ * 2);
  unsigned short* Qb     = (unsigned short*)alloc(MX * D_ * 2);
  unsigned short* Kb     = (unsigned short*)alloc(MX * D_ * 2);
  unsigned short* Vtb    = (unsigned short*)alloc((long)D_ * MX * 2);     // [768][8192]
  unsigned short* ctxb   = (unsigned short*)alloc(MX * D_ * 2);
  unsigned short* attnb  = (unsigned short*)alloc(MX * D_ * 2);
  unsigned short* interb = (unsigned short*)alloc(MX * FF_ * 2);
  unsigned short* WqT    = (unsigned short*)alloc((long)L_ * D_ * D_ * 2);
  unsigned short* WkT    = (unsigned short*)alloc((long)L_ * D_ * D_ * 2);
  unsigned short* WvT    = (unsigned short*)alloc((long)L_ * D_ * D_ * 2);
  unsigned short* WoT    = (unsigned short*)alloc((long)L_ * D_ * D_ * 2);
  unsigned short* WiT    = (unsigned short*)alloc((long)L_ * D_ * FF_ * 2);
  unsigned short* WfT    = (unsigned short*)alloc((long)L_ * D_ * FF_ * 2);
  float*          scores = (float*)alloc(NS * 4);
  unsigned short* probsB = (unsigned short*)alloc(NS * 2);
  float*          tmpf   = (float*)alloc(MX * D_ * 4);
  float*          attnf  = (float*)alloc(MX * D_ * 4);
  float*          red    = (float*)alloc(256);

  float* xOutF  = (float*)d_out;                       // final x, also residual
  float* probsF = (float*)d_out + MX * (long)D_;       // [B,H,S1,S2]

  auto gemm = [&](const unsigned short* A, const unsigned short* Bt,
                  const float* bias, float* Cf, unsigned short* Cb,
                  int M, int N, int K, int lda, int ldb, int ldc,
                  long aSB, long aSH, long bSB, long bSH, long cSB, long cSH,
                  int nH, int Z, float scale, int epi, int ldcT, float* redOut) {
    dim3 g(N / 64, (M + 255) / 256, Z);
    wmma_gemm_kernel<<<g, dim3(256), 0, stream>>>(
        A, Bt, bias, Cf, Cb, M, N, K, lda, ldb, ldc,
        aSB, aSH, bSB, bSH, cSB, cSH, nH, scale, epi, ldcT, redOut);
  };

  // ---- one-time (per launch) conversions ----
  {
    long n = MX * D_;
    cvt_f32_bf16_kernel<<<(n + 255) / 256, 256, 0, stream>>>(Xin, Xb, n);
    cvt_f32_bf16_kernel<<<(n + 255) / 256, 256, 0, stream>>>(Yin, Yb, n);
    dim3 tb(32, 8);
    transpose_cvt_kernel<<<dim3(D_ / 32,  D_ / 32, L_), tb, 0, stream>>>(Wq, WqT, D_, D_);
    transpose_cvt_kernel<<<dim3(D_ / 32,  D_ / 32, L_), tb, 0, stream>>>(Wk, WkT, D_, D_);
    transpose_cvt_kernel<<<dim3(D_ / 32,  D_ / 32, L_), tb, 0, stream>>>(Wv, WvT, D_, D_);
    transpose_cvt_kernel<<<dim3(D_ / 32,  D_ / 32, L_), tb, 0, stream>>>(Wo, WoT, D_, D_);
    transpose_cvt_kernel<<<dim3(FF_ / 32, D_ / 32, L_), tb, 0, stream>>>(Wi, WiT, D_, FF_);
    transpose_cvt_kernel<<<dim3(D_ / 32, FF_ / 32, L_), tb, 0, stream>>>(Wf, WfT, FF_, D_);
  }

  for (int i = 0; i < L_; ++i) {
    const long wOff  = (long)i * D_ * D_;
    const long wOffF = (long)i * D_ * FF_;

    // Q = Xb * WqT + bq (bf16)                 K = Yb * WkT + bk (bf16)
    gemm(Xb, WqT + wOff, bq + i * D_, nullptr, Qb, MX, D_, D_, D_, D_, D_,
         0, 0, 0, 0, 0, 0, 1, 1, 1.0f, 2, 0, nullptr);
    gemm(Yb, WkT + wOff, bk + i * D_, nullptr, Kb, MX, D_, D_, D_, D_, D_,
         0, 0, 0, 0, 0, 0, 1, 1, 1.0f, 2, 0, nullptr);
    // V projection, stored TRANSPOSED -> Vtb[h*64+d][b*512+s]
    gemm(Yb, WvT + wOff, bv + i * D_, nullptr, Vtb, MX, D_, D_, D_, D_, D_,
         0, 0, 0, 0, 0, 0, 1, 1, 1.0f, 8, (int)MX, nullptr);

    // scores[b,h] = Q_head * K_head^T / 8 (fp32 NT out) with FUSED global
    // sum/sumsq reduction into red[0..1]; batched over 192 (b,h) pairs.
    zero2_kernel<<<1, 32, 0, stream>>>(red);
    gemm(Qb, Kb, nullptr, scores, nullptr, S1_, S2_, DH_, D_, D_, S2_,
         (long)S1_ * D_, DH_, (long)S2_ * D_, DH_,
         (long)H_ * S1_ * S2_, (long)S1_ * S2_,
         H_, B_ * H_, 0.125f, 1 | 16 | 32, 0, red);

    softmax_kernel<<<(unsigned)(B_ * H_ * S1_), 256, 0, stream>>>(
        scores, red, (float)NS, probsF, probsB);

    // ctx[b,h] = probs * V  (B operand = Vtb with ldb = 8192), bf16 out
    gemm(probsB, Vtb, nullptr, nullptr, ctxb, S1_, DH_, S2_, S2_, (int)MX, D_,
         (long)H_ * S1_ * S2_, (long)S1_ * S2_,
         (long)S2_, (long)DH_ * MX,
         (long)S1_ * D_, DH_,
         H_, B_ * H_, 1.0f, 2, 0, nullptr);

    // attn pre-LN = ctx * WoT + bo  (fp32)
    gemm(ctxb, WoT + wOff, bo + i * D_, tmpf, nullptr, MX, D_, D_, D_, D_, D_,
         0, 0, 0, 0, 0, 0, 1, 1, 1.0f, 1, 0, nullptr);
    // LN1(tmp + x_resid) -> attnf (fp32) + attnb (bf16)
    ln_residual_kernel<<<(unsigned)MX, 256, 0, stream>>>(
        tmpf, (i == 0) ? Xin : xOutF, g1 + i * D_, be1 + i * D_, attnf, attnb);

    // inter = gelu(attn * WiT + bi)  (bf16)
    gemm(attnb, WiT + wOffF, bi + i * FF_, nullptr, interb, MX, FF_, D_, D_, D_, FF_,
         0, 0, 0, 0, 0, 0, 1, 1, 1.0f, 2 | 4, 0, nullptr);
    // ffn = inter * WfT + bff  (fp32)
    gemm(interb, WfT + wOffF, bf + i * D_, tmpf, nullptr, MX, D_, FF_, FF_, FF_, D_,
         0, 0, 0, 0, 0, 0, 1, 1, 1.0f, 1, 0, nullptr);
    // x = LN2(ffn + attn) -> d_out (fp32, next-layer residual) + Xb (bf16)
    ln_residual_kernel<<<(unsigned)MX, 256, 0, stream>>>(
        tmpf, attnf, g2 + i * D_, be2 + i * D_, xOutF, Xb);
  }
}
